// Encoder_2886218023681
// MI455X (gfx1250) — matrix operand order repaired and checked by reading, not compiled, not run on hardware
//
#include <hip/hip_runtime.h>
#include <cstdint>
#include <cstddef>

// ---- problem constants (match reference) ----
#define B_      16
#define T_TOT   48
#define HIST    24
#define NN      256
#define FF      8
#define EEDGE   4096
#define EMBD    32
#define HIDD    64
#define HEADS   4
#define DH      16
#define IN_CONV 40          // F-1 + EMB + 1
#define IN_FC   105         // F + 1 + EMB + HID
#define KP_CONV 64          // IN_CONV padded to mult of 32
#define KP_FC   128         // IN_FC padded to mult of 32

#define BT      (B_*HIST)        // 384
#define RTOT    (BT*NN)          // 98304
#define BTE     ((long)BT*EEDGE) // 1572864

typedef __attribute__((ext_vector_type(16))) _Float16 v16h;
typedef __attribute__((ext_vector_type(8)))  _Float16 v8h;
typedef __attribute__((ext_vector_type(8)))  float    v8f;
typedef __attribute__((ext_vector_type(4)))  int      v4i_;

#if __has_builtin(__builtin_amdgcn_global_load_async_to_lds_b128)
#define HAVE_ASYNC_LDS 1
typedef __attribute__((address_space(1))) v4i_ gbl_v4i;   // global b128 pointee
typedef __attribute__((address_space(3))) v4i_ lds_v4i;   // LDS b128 pointee
#endif

static __device__ __forceinline__ void wait_async0() {
#if __has_builtin(__builtin_amdgcn_s_wait_asynccnt)
  __builtin_amdgcn_s_wait_asynccnt(0);
#else
  asm volatile("s_wait_asynccnt 0" ::: "memory");
#endif
}

static __device__ __forceinline__ void atomicMaxF(float* addr, float val) {
  // monotonic int/uint mapping trick; valid with -inf init
  if (val >= 0.0f) atomicMax((int*)addr, __float_as_int(val));
  else             atomicMin((unsigned int*)addr, __float_as_uint(val));
}

// ---------------------------------------------------------------------------
// K1: per-node features -> xh (f16, [R, 64], zero padded) + speed/wdir
// ---------------------------------------------------------------------------
__global__ void k_features(const float* __restrict__ X, const float* __restrict__ Y,
                           const float* __restrict__ pos_table,
                           const float* __restrict__ spt_table,
                           _Float16* __restrict__ xh,
                           float* __restrict__ spd, float* __restrict__ wdr) {
  long gid = (long)blockIdx.x * blockDim.x + threadIdx.x;
  if (gid >= (long)RTOT * KP_CONV) return;
  int r  = (int)(gid >> 6);
  int j  = (int)(gid & 63);
  int n  = r % NN;
  int bt = r / NN;
  int t  = bt % HIST;
  int b  = bt / HIST;
  long xbase = (((long)(b * T_TOT + t)) * NN + n) * FF;
  float v = 0.0f;
  if (j < EMBD) {
    int idx = (int)X[xbase + FF - 1];
    v = pos_table[t * EMBD + j] + spt_table[idx * EMBD + j];
  } else if (j == EMBD) {
    v = Y[((long)(b * T_TOT + t)) * NN + n];
  } else if (j < EMBD + 1 + FF - 1) {
    v = X[xbase + (j - EMBD - 1)];
  }
  xh[gid] = (_Float16)v;
  if (j == 0) {
    float u10 = X[xbase + 5] * 3.0f + 0.5f;   // U10_STD, U10_MEAN
    float v10 = X[xbase + 6] * 3.0f + 0.3f;   // V10_STD, V10_MEAN
    spd[r] = sqrtf(u10 * u10 + v10 * v10);
    float a = 1.5707963267948966f - atan2f(-v10, -u10);
    const float twopi = 6.283185307179586f;
    wdr[r] = a - floorf(a / twopi) * twopi;   // jnp.mod semantics
  }
}

// ---------------------------------------------------------------------------
// K2: weights -> f16, TRANSPOSED to [N, K] so WMMA B-fragments are two
// contiguous 16B vector loads per lane.
// ---------------------------------------------------------------------------
__global__ void k_prep_weights(
    const float* Wq, const float* bq, const float* Wk, const float* bk,
    const float* Wv, const float* bv, const float* Wskip, const float* bskip,
    const float* Wfc, const float* Waq, const float* baq,
    const float* Wak, const float* bak, const float* Wav, const float* bav,
    const float* Wao, const float* Wmlp,
    _Float16* Wqkvs_t, float* bqkvs,
    _Float16* Wfc_t, _Float16* Wa_t, float* ba,
    _Float16* Wao_t, _Float16* Wmlp_t) {
  int tid = blockIdx.x * blockDim.x + threadIdx.x;
  int nth = gridDim.x * blockDim.x;
  // Wqkvs_t: [256, 64]  (n, k) = [Wq|Wk|Wv|Wskip]^T, K padded 40->64
  for (int i = tid; i < 256 * KP_CONV; i += nth) {
    int n = i >> 6, kk = i & 63;
    int grp = n >> 6, j = n & 63;
    float w = 0.0f;
    if (kk < IN_CONV) {
      const float* Ws = grp == 0 ? Wq : grp == 1 ? Wk : grp == 2 ? Wv : Wskip;
      w = Ws[kk * HIDD + j];
    }
    Wqkvs_t[i] = (_Float16)w;
  }
  for (int c = tid; c < 256; c += nth) {
    int grp = c >> 6, j = c & 63;
    const float* bs = grp == 0 ? bq : grp == 1 ? bk : grp == 2 ? bv : bskip;
    bqkvs[c] = bs[j];
  }
  // Wfc_t: [64, 128], K padded 105->128
  for (int i = tid; i < 64 * KP_FC; i += nth) {
    int n = i >> 7, kk = i & 127;
    Wfc_t[i] = (_Float16)(kk < IN_FC ? Wfc[kk * HIDD + n] : 0.0f);
  }
  // Wa_t: [192, 64] = [Waq|Wak|Wav]^T
  for (int i = tid; i < 192 * HIDD; i += nth) {
    int n = i >> 6, kk = i & 63;
    int grp = n >> 6, j = n & 63;
    const float* Ws = grp == 0 ? Waq : grp == 1 ? Wak : Wav;
    Wa_t[i] = (_Float16)Ws[kk * HIDD + j];
  }
  for (int c = tid; c < 192; c += nth) {
    int grp = c >> 6, j = c & 63;
    ba[c] = (grp == 0 ? baq : grp == 1 ? bak : bav)[j];
  }
  // Wao_t / Wmlp_t: [64, 64] transposed
  for (int i = tid; i < 64 * 64; i += nth) {
    int n = i >> 6, kk = i & 63;
    Wao_t[i]  = (_Float16)Wao[kk * HIDD + n];
    Wmlp_t[i] = (_Float16)Wmlp[kk * HIDD + n];
  }
}

// ---------------------------------------------------------------------------
// Generic WMMA GEMM: C[M,N] = A[M,K](f16,row) @ B (given as Bt[N,K] f16) + bias
// One wave per 16x64 output (4 WMMA tiles sharing one A fragment).
// K is a template constant (fully unrolled, loads hoisted over WMMAs).
// M mult of 16, N mult of 64, K mult of 32.
// ---------------------------------------------------------------------------
template <int K>
__global__ void __launch_bounds__(128) k_wmma_gemm(
    const _Float16* __restrict__ A, const _Float16* __restrict__ Bt,
    const float* __restrict__ bias,
    float* __restrict__ C32, _Float16* __restrict__ C16,
    int M, int N) {
  int wave = blockIdx.x * (blockDim.x >> 5) + (int)(threadIdx.x >> 5);
  int ngrp = N >> 6;                    // groups of 64 output columns
  int mt = wave / ngrp;
  int ng = wave - mt * ngrp;
  if ((mt << 4) >= M) return;
  int lane = threadIdx.x & 31;
  int row  = lane & 15;    // A row within tile / B,C column within tile
  int hi   = lane >> 4;    // lane half
  v8f acc[4] = {};
  const _Float16* arow = A + (size_t)((mt << 4) + row) * K;
  const _Float16* brow0 = Bt + (size_t)((ng << 6) + row) * K + (hi << 4);
#pragma unroll
  for (int k0 = 0; k0 < K; k0 += 32) {
    // A frag: elem i -> K = k0 + hi*16 + i (contiguous)
    v8h a0 = *(const v8h*)(arow + k0 + (hi << 4));
    v8h a1 = *(const v8h*)(arow + k0 + (hi << 4) + 8);
    v16h a;
#pragma unroll
    for (int i = 0; i < 8; ++i) { a[i] = a0[i]; a[i + 8] = a1[i]; }
#pragma unroll
    for (int u = 0; u < 4; ++u) {
      // B frag: lane holds column, elem i -> K = k0 + hi*16 + i (contiguous in Bt)
      const _Float16* brow = brow0 + (size_t)(u << 4) * K + k0;
      v8h b0 = *(const v8h*)(brow);
      v8h b1 = *(const v8h*)(brow + 8);
      v16h bf;
#pragma unroll
      for (int i = 0; i < 8; ++i) { bf[i] = b0[i]; bf[i + 8] = b1[i]; }
      acc[u] = __builtin_amdgcn_wmma_f32_16x16x32_f16(false, a, false, bf,
                                                      (short)0, acc[u], false, false);
    }
  }
#pragma unroll
  for (int u = 0; u < 4; ++u) {
#pragma unroll
    for (int rr = 0; rr < 8; ++rr) {
      int m = (mt << 4) + (hi << 3) + rr;
      int n = (ng << 6) + (u << 4) + row;
      float v = acc[u][rr] + (bias ? bias[n] : 0.0f);
      if (C32) C32[(size_t)m * N + n] = v;
      if (C16) C16[(size_t)m * N + n] = (_Float16)v;
    }
  }
}

// ---------------------------------------------------------------------------
// K4: init segment-softmax state (every call — graph replay safe)
// ---------------------------------------------------------------------------
__global__ void k_init_seg(float* __restrict__ m, float* __restrict__ den,
                           float* __restrict__ agg) {
  long gid = (long)blockIdx.x * blockDim.x + threadIdx.x;
  if (gid < (long)BT * NN) { m[gid] = -__builtin_inff(); den[gid] = 0.0f; }
  if (gid < (long)RTOT * HIDD) agg[gid] = 0.0f;
}

// ---------------------------------------------------------------------------
// K5: edge pass 1 — edge attrs, logits, atomic segment-max over dst
// ---------------------------------------------------------------------------
__global__ void k_edge_logits(const int* __restrict__ eidx,
                              const float* __restrict__ eastat,
                              const float* __restrict__ We,
                              const float* __restrict__ spd,
                              const float* __restrict__ wdr,
                              const float* __restrict__ qkvs,
                              float* __restrict__ eattr,
                              float* __restrict__ logits,
                              float* __restrict__ mbuf) {
  long gid = (long)blockIdx.x * blockDim.x + threadIdx.x;
  if (gid >= BTE) return;
  int e  = (int)(gid % EEDGE);
  int bt = (int)(gid / EEDGE);
  int src = eidx[e], dst = eidx[EEDGE + e];
  float sp = spd[bt * NN + src], di = wdr[bt * NN + src];
  float d0 = eastat[e * 2], d1 = eastat[e * 2 + 1];
  float theta = fabsf(d1 - di);
  float adv = fmaxf(0.0f, 3.0f * sp * cosf(theta) / d0);
  float ea[5] = { d0, d1, sp, di, adv };
#pragma unroll
  for (int c = 0; c < 5; ++c) eattr[gid * 5 + c] = ea[c];
  const float* qrow = qkvs + (size_t)(bt * NN + dst) * 256;          // q block
  const float* krow = qkvs + (size_t)(bt * NN + src) * 256 + 64;     // k block
  float dot = 0.0f;
  for (int j = 0; j < HIDD; ++j) {
    float eej = ea[0]*We[j] + ea[1]*We[64+j] + ea[2]*We[128+j]
              + ea[3]*We[192+j] + ea[4]*We[256+j];
    dot += qrow[j] * (krow[j] + eej);
  }
  float lg = dot * 0.125f;   // / sqrt(HID)
  logits[gid] = lg;
  atomicMaxF(&mbuf[bt * NN + dst], lg);
}

// ---------------------------------------------------------------------------
// K6: edge pass 2 — exp, atomic den / weighted-value accumulation
// ---------------------------------------------------------------------------
__global__ void k_edge_agg(const int* __restrict__ eidx,
                           const float* __restrict__ We,
                           const float* __restrict__ eattr,
                           const float* __restrict__ logits,
                           const float* __restrict__ mbuf,
                           const float* __restrict__ qkvs,
                           float* __restrict__ den, float* __restrict__ agg) {
  long gid = (long)blockIdx.x * blockDim.x + threadIdx.x;
  if (gid >= BTE * HIDD) return;
  long eg = gid >> 6;
  int  j  = (int)(gid & 63);
  int  e  = (int)(eg % EEDGE);
  int  bt = (int)(eg / EEDGE);
  int src = eidx[e], dst = eidx[EEDGE + e];
  float a = expf(logits[eg] - mbuf[bt * NN + dst]);
  const float* ea = eattr + eg * 5;
  float eej = ea[0]*We[j] + ea[1]*We[64+j] + ea[2]*We[128+j]
            + ea[3]*We[192+j] + ea[4]*We[256+j];
  float val = qkvs[(size_t)(bt * NN + src) * 256 + 128 + j] + eej;   // v block
  atomicAdd(&agg[(size_t)(bt * NN + dst) * HIDD + j], a * val);
  if (j == 0) atomicAdd(&den[bt * NN + dst], a);
}

// ---------------------------------------------------------------------------
// K7: word = sigmoid(agg/den + skip + bskip); build cat (f16, [R,128] padded)
// ---------------------------------------------------------------------------
__global__ void k_word_cat(const float* __restrict__ X, const float* __restrict__ Y,
                           const float* __restrict__ pos_table,
                           const float* __restrict__ agg, const float* __restrict__ den,
                           const float* __restrict__ qkvs, const float* __restrict__ bskip,
                           _Float16* __restrict__ cath) {
  long gid = (long)blockIdx.x * blockDim.x + threadIdx.x;
  if (gid >= (long)RTOT * KP_FC) return;
  int r = (int)(gid >> 7);
  int j = (int)(gid & 127);
  int n = r % NN;
  int bt = r / NN;
  int t = bt % HIST;
  int b = bt / HIST;
  float v = 0.0f;
  if (j < EMBD) {
    v = pos_table[t * EMBD + j];
  } else if (j < EMBD + HIDD) {
    int j2 = j - EMBD;
    float w = agg[(size_t)r * HIDD + j2] / den[r]
            + qkvs[(size_t)r * 256 + 192 + j2] + bskip[j2];   // skip block
    v = 1.0f / (1.0f + expf(-w));
  } else if (j < EMBD + HIDD + FF) {
    v = X[(((long)(b * T_TOT + t)) * NN + n) * FF + (j - EMBD - HIDD)];
  } else if (j == EMBD + HIDD + FF) {
    v = Y[((long)(b * T_TOT + t)) * NN + n];
  }
  cath[gid] = (_Float16)v;
}

// ---------------------------------------------------------------------------
// K7b: vT[bt][h][d][n] = va[bt,n,h,d]  (transpose V for contiguous B-fragments)
// ---------------------------------------------------------------------------
__global__ void k_transpose_v(const _Float16* __restrict__ qkva,
                              _Float16* __restrict__ vT) {
  long gid = (long)blockIdx.x * blockDim.x + threadIdx.x;
  if (gid >= (long)BT * HEADS * DH * NN) return;
  int n  = (int)(gid & 255);
  int d  = (int)((gid >> 8) & 15);
  int h  = (int)((gid >> 12) & 3);
  int bt = (int)(gid >> 14);
  vT[gid] = qkva[((size_t)bt * NN + n) * 192 + 128 + h * DH + d];
}

// ---------------------------------------------------------------------------
// K7c: kpad[bt][h][n][32] = ka (Dh=16 real + 16 zero pad) -> score B-fragment
// becomes two plain b128 loads for BOTH lane halves (no per-element selects).
// ---------------------------------------------------------------------------
__global__ void k_pad_k(const _Float16* __restrict__ qkva,
                        _Float16* __restrict__ kpad) {
  long gid = (long)blockIdx.x * blockDim.x + threadIdx.x;
  if (gid >= (long)BT * HEADS * NN * 32) return;
  int kk = (int)(gid & 31);
  int n  = (int)((gid >> 5) & 255);
  int h  = (int)((gid >> 13) & 3);
  int bt = (int)(gid >> 15);
  _Float16 v = (_Float16)0.0f;
  if (kk < DH) v = qkva[((size_t)bt * NN + n) * 192 + 64 + h * DH + kk];
  kpad[gid] = v;
}

// ---------------------------------------------------------------------------
// K8: dense per-(b,t,head) node attention via WMMA.
// One wave per (b,t,head, 16-query strip).
// V-head strip (8KB) is staged into LDS with async-to-LDS DMA (tracked by
// ASYNCcnt), overlapping the 16 score WMMAs + softmax; s_wait_asynccnt 0
// before the attn@V phase.
// ---------------------------------------------------------------------------
__global__ void __launch_bounds__(32) k_node_attn(const _Float16* __restrict__ qkva,
                                                  const _Float16* __restrict__ kpad,
                                                  const _Float16* __restrict__ vT,
                                                  _Float16* __restrict__ ctx16) {
  __shared__ float    sc[16 * 256];
  __shared__ _Float16 at[16 * 256];
  __shared__ _Float16 vlds[16 * 256];
  int bid = blockIdx.x;
  int s   = bid & 15;          // query strip
  int h   = (bid >> 4) & 3;    // head
  int bt  = bid >> 6;
  int lane = threadIdx.x;
  int row  = lane & 15;
  int hi   = lane >> 4;
  size_t base = (size_t)bt * NN;
  const _Float16* vTh = vT + ((size_t)(bt * HEADS + h)) * DH * NN;
  const _Float16* kph = kpad + ((size_t)(bt * HEADS + h)) * NN * 32;

#ifdef HAVE_ASYNC_LDS
  // kick off async DMA of the V strip (16x256 f16 = 8KB) into LDS;
  // completes while we do scores + softmax.
#pragma unroll
  for (int it = 0; it < 16; ++it) {
    int idx = (it * 32 + lane) * 8;                 // 8 halves = 16B per op
    __builtin_amdgcn_global_load_async_to_lds_b128(
        (gbl_v4i*)(vTh + idx), (lds_v4i*)(vlds + idx), 0, 0);
  }
#else
  // fall back: prefetch this wave's V strip (used after softmax)
  __builtin_prefetch(vTh + (size_t)row * NN, 0, 3);
#endif

  // A fragment: queries; valid half = 8 contiguous halves at offset hi*8
  v16h a;
  {
    v8h q0 = *(const v8h*)(qkva + (base + s * 16 + row) * 192 + h * DH + (hi << 3));
#pragma unroll
    for (int i = 0; i < 8; ++i) { a[i] = q0[i]; a[i + 8] = (_Float16)0.0f; }
  }
  // scores strip: 16 key tiles; B frag elems 0..7 -> kk = hi*8+i, elems 8..15 -> +16 (zero pad)
  for (int mt = 0; mt < 16; ++mt) {
    const _Float16* kp = kph + (size_t)(mt * 16 + row) * 32 + (hi << 3);
    v8h b0 = *(const v8h*)(kp);
    v8h b1 = *(const v8h*)(kp + 16);
    v16h bf;
#pragma unroll
    for (int i = 0; i < 8; ++i) { bf[i] = b0[i]; bf[i + 8] = b1[i]; }
    v8f c = {};
    c = __builtin_amdgcn_wmma_f32_16x16x32_f16(false, a, false, bf,
                                               (short)0, c, false, false);
#pragma unroll
    for (int rr = 0; rr < 8; ++rr)
      sc[((hi << 3) + rr) * 256 + mt * 16 + row] = c[rr] * 0.25f;  // / sqrt(Dh)
  }
  __syncthreads();

  // softmax over 256 keys; lane: row = lane&15, column half = hi
  float mx = -__builtin_inff();
  for (int nc = hi * 128; nc < hi * 128 + 128; ++nc)
    mx = fmaxf(mx, sc[row * 256 + nc]);
  mx = fmaxf(mx, __shfl_xor(mx, 16));
  float sum = 0.0f;
  for (int nc = hi * 128; nc < hi * 128 + 128; ++nc)
    sum += expf(sc[row * 256 + nc] - mx);
  sum += __shfl_xor(sum, 16);
  float inv = 1.0f / sum;
  for (int nc = hi * 128; nc < hi * 128 + 128; ++nc)
    at[row * 256 + nc] = (_Float16)(expf(sc[row * 256 + nc] - mx) * inv);

#ifdef HAVE_ASYNC_LDS
  wait_async0();               // V strip resident in LDS
#endif
  __syncthreads();

  // ctx = attn[16,256] @ V[256,16] : 8 K-chunks of 32
  v8f acc = {};
  for (int kc = 0; kc < 8; ++kc) {
    // A2 from LDS: elem i at kc*32+hi*16+i (two b128 reads)
    v8h a0 = *(const v8h*)(at + row * 256 + kc * 32 + (hi << 4));
    v8h a1 = *(const v8h*)(at + row * 256 + kc * 32 + (hi << 4) + 8);
    v16h a2;
#pragma unroll
    for (int i = 0; i < 8; ++i) { a2[i] = a0[i]; a2[i + 8] = a1[i]; }
    // B2: lane col = row (dh out); elem i -> node kc*32 + hi*16 + i
#ifdef HAVE_ASYNC_LDS
    const _Float16* vp = vlds + (size_t)row * NN + kc * 32 + (hi << 4);
#else
    const _Float16* vp = vTh + (size_t)row * NN + kc * 32 + (hi << 4);
#endif
    v8h b0 = *(const v8h*)(vp);
    v8h b1 = *(const v8h*)(vp + 8);
    v16h b2;
#pragma unroll
    for (int i = 0; i < 8; ++i) { b2[i] = b0[i]; b2[i + 8] = b1[i]; }
    acc = __builtin_amdgcn_wmma_f32_16x16x32_f16(false, a2, false, b2,
                                                 (short)0, acc, false, false);
  }
#pragma unroll
  for (int rr = 0; rr < 8; ++rr) {
    int m = s * 16 + (hi << 3) + rr;
    ctx16[(base + m) * 64 + h * DH + row] = (_Float16)acc[rr];
  }
}

// ---------------------------------------------------------------------------
extern "C" void kernel_launch(void* const* d_in, const int* in_sizes, int n_in,
                              void* d_out, int out_size, void* d_ws, size_t ws_size,
                              hipStream_t stream) {
  (void)in_sizes; (void)n_in; (void)out_size; (void)ws_size;
  const float* X         = (const float*)d_in[0];
  const float* Y         = (const float*)d_in[1];
  const float* pos_table = (const float*)d_in[2];
  const float* spt_table = (const float*)d_in[3];
  const float* Wq    = (const float*)d_in[4];  const float* bq    = (const float*)d_in[5];
  const float* Wk    = (const float*)d_in[6];  const float* bk    = (const float*)d_in[7];
  const float* Wv    = (const float*)d_in[8];  const float* bv    = (const float*)d_in[9];
  const float* We    = (const float*)d_in[10];
  const float* Wskip = (const float*)d_in[11]; const float* bskip = (const float*)d_in[12];
  const float* Wfc   = (const float*)d_in[13]; const float* bfc   = (const float*)d_in[14];
  const float* Waq   = (const float*)d_in[15]; const float* baq   = (const float*)d_in[16];
  const float* Wak   = (const float*)d_in[17]; const float* bak   = (const float*)d_in[18];
  const float* Wav   = (const float*)d_in[19]; const float* bav   = (const float*)d_in[20];
  const float* Wao   = (const float*)d_in[21]; const float* bao   = (const float*)d_in[22];
  const float* Wmlp  = (const float*)d_in[23]; const float* bmlp  = (const float*)d_in[24];
  const int*   eidx  = (const int*)d_in[25];
  const float* eastat= (const float*)d_in[26];

  char* ws = (char*)d_ws;
  size_t off = 0;
  auto alloc = [&](size_t bytes) -> char* {
    char* p = ws + off;
    off = (off + bytes + 255) & ~(size_t)255;
    return p;
  };
  _Float16* xh     = (_Float16*)alloc((size_t)RTOT * KP_CONV * 2);
  float*    qkvs   = (float*)   alloc((size_t)RTOT * 256 * 4);
  float*    spd    = (float*)   alloc((size_t)BT * NN * 4);
  float*    wdr    = (float*)   alloc((size_t)BT * NN * 4);
  float*    eattr  = (float*)   alloc((size_t)BTE * 5 * 4);
  float*    logits = (float*)   alloc((size_t)BTE * 4);
  float*    mbuf   = (float*)   alloc((size_t)BT * NN * 4);
  float*    den    = (float*)   alloc((size_t)BT * NN * 4);
  float*    agg    = (float*)   alloc((size_t)RTOT * HIDD * 4);
  _Float16* cath   = (_Float16*)alloc((size_t)RTOT * KP_FC * 2);
  _Float16* h16    = (_Float16*)alloc((size_t)RTOT * HIDD * 2);
  _Float16* qkva16 = (_Float16*)alloc((size_t)RTOT * 192 * 2);
  _Float16* vT     = (_Float16*)alloc((size_t)BT * HEADS * DH * NN * 2);
  _Float16* kpad   = (_Float16*)alloc((size_t)BT * HEADS * NN * 32 * 2);
  _Float16* ctx16  = (_Float16*)alloc((size_t)RTOT * HIDD * 2);
  _Float16* ao16   = (_Float16*)alloc((size_t)RTOT * HIDD * 2);
  _Float16* Wqkvs_t= (_Float16*)alloc(256 * KP_CONV * 2);
  float*    bqkvs  = (float*)   alloc(256 * 4);
  _Float16* Wfc_t  = (_Float16*)alloc(64 * KP_FC * 2);
  _Float16* Wa_t   = (_Float16*)alloc(192 * HIDD * 2);
  float*    ba     = (float*)   alloc(192 * 4);
  _Float16* Wao_t  = (_Float16*)alloc(64 * 64 * 2);
  _Float16* Wmlp_t = (_Float16*)alloc(64 * 64 * 2);

  const int thr = 256;
  // 1. per-node features
  {
    long n = (long)RTOT * KP_CONV;
    k_features<<<(unsigned)((n + thr - 1) / thr), thr, 0, stream>>>(
        X, Y, pos_table, spt_table, xh, spd, wdr);
  }
  // 2. weights -> f16 transposed
  k_prep_weights<<<64, 256, 0, stream>>>(
      Wq, bq, Wk, bk, Wv, bv, Wskip, bskip, Wfc, Waq, baq, Wak, bak, Wav, bav,
      Wao, Wmlp, Wqkvs_t, bqkvs, Wfc_t, Wa_t, ba, Wao_t, Wmlp_t);
  // 3. fused q|k|v|skip GEMM: [R,64] @ [64,256]
  {
    long waves = (long)(RTOT / 16) * (256 / 64);
    k_wmma_gemm<KP_CONV><<<(unsigned)((waves + 3) / 4), 128, 0, stream>>>(
        xh, Wqkvs_t, bqkvs, qkvs, nullptr, RTOT, 256);
  }
  // 4. init segment state
  {
    long n = (long)RTOT * HIDD;
    k_init_seg<<<(unsigned)((n + thr - 1) / thr), thr, 0, stream>>>(mbuf, den, agg);
  }
  // 5. edge pass 1: logits + segment max
  k_edge_logits<<<(unsigned)((BTE + thr - 1) / thr), thr, 0, stream>>>(
      eidx, eastat, We, spd, wdr, qkvs, eattr, logits, mbuf);
  // 6. edge pass 2: weighted aggregation
  {
    long n = BTE * HIDD;
    k_edge_agg<<<(unsigned)((n + thr - 1) / thr), thr, 0, stream>>>(
        eidx, We, eattr, logits, mbuf, qkvs, den, agg);
  }
  // 7. word + cat
  {
    long n = (long)RTOT * KP_FC;
    k_word_cat<<<(unsigned)((n + thr - 1) / thr), thr, 0, stream>>>(
        X, Y, pos_table, agg, den, qkvs, bskip, cath);
  }
  // 8. fc GEMM: [R,128] @ [128,64] -> h16
  {
    long waves = (long)(RTOT / 16) * (64 / 64);
    k_wmma_gemm<KP_FC><<<(unsigned)((waves + 3) / 4), 128, 0, stream>>>(
        cath, Wfc_t, bfc, nullptr, h16, RTOT, 64);
  }
  // 9. fused qa|ka|va GEMM: [R,64] @ [64,192] -> qkva16
  {
    long waves = (long)(RTOT / 16) * (192 / 64);
    k_wmma_gemm<HIDD><<<(unsigned)((waves + 3) / 4), 128, 0, stream>>>(
        h16, Wa_t, ba, nullptr, qkva16, RTOT, 192);
  }
  // 9b. transpose V / zero-pad K per (bt, head)
  {
    long n = (long)BT * HEADS * DH * NN;
    k_transpose_v<<<(unsigned)((n + thr - 1) / thr), thr, 0, stream>>>(qkva16, vT);
    long n2 = (long)BT * HEADS * NN * 32;
    k_pad_k<<<(unsigned)((n2 + thr - 1) / thr), thr, 0, stream>>>(qkva16, kpad);
  }
  // 10. dense node attention
  k_node_attn<<<BT * HEADS * 16, 32, 0, stream>>>(qkva16, kpad, vT, ctx16);
  // 11. out projection: ctx @ Wao + bao -> ao16
  {
    long waves = (long)(RTOT / 16) * (64 / 64);
    k_wmma_gemm<HIDD><<<(unsigned)((waves + 3) / 4), 128, 0, stream>>>(
        ctx16, Wao_t, bao, nullptr, ao16, RTOT, 64);
  }
  // 12. mlp: ao @ Wmlp + bmlp -> d_out (f32)
  {
    long waves = (long)(RTOT / 16) * (64 / 64);
    k_wmma_gemm<HIDD><<<(unsigned)((waves + 3) / 4), 128, 0, stream>>>(
        ao16, Wmlp_t, bmlp, (float*)d_out, nullptr, RTOT, 64);
  }
}
